// SimpleAttention_6717328851259
// MI455X (gfx1250) — matrix operand order repaired and checked by reading, not compiled, not run on hardware
//
#include <hip/hip_runtime.h>
#include <hip/hip_bf16.h>

// ---------------------------------------------------------------------------
// SimpleAttention diagonal-softmax for MI455X (gfx1250, wave32, WMMA)
//
// pass 1: qk = x @ W^T + b   (M=8192, F=1536, K=768)  -> Q,K in f16 [B,H,N,D]
// pass 2: per (b,h): S = Q K^T * 0.125, streaming online softmax keeping only
//         rowwise (max, sumexp) + the diagonal logit   -> diag[B*H, N]
//         K chunks streamed via GLOBAL_LOAD_ASYNC_TO_LDS_B128 (double-buffer)
// pass 3: mean over heads -> out[B, N]
// ---------------------------------------------------------------------------

typedef __attribute__((ext_vector_type(16))) _Float16 v16h;
typedef __attribute__((ext_vector_type(8)))  _Float16 v8h;
typedef __attribute__((ext_vector_type(4)))  _Float16 v4h;
typedef __attribute__((ext_vector_type(8)))  float    v8f;

#define DIMC     768
#define FEATS    1536
#define NHEAD    12
#define HDIM     64
#define SEQN     2048
#define BATCH    4
#define MROWS    (BATCH * SEQN)
#define QK_HALFS ((size_t)BATCH * NHEAD * SEQN * HDIM)

#if __has_builtin(__builtin_amdgcn_global_load_async_to_lds_b128)
#define ASYNC_LDS 1
// builtin signature (from hipcc diagnostic): param0 = AS(1) int4*, lds = AS(3)
typedef int v4i __attribute__((vector_size(16)));
typedef __attribute__((address_space(1))) v4i gv4i;
typedef __attribute__((address_space(3))) v4i lv4i;
#else
#define ASYNC_LDS 0
#endif

__device__ __forceinline__ void wait_async0() {
#if ASYNC_LDS
#if __has_builtin(__builtin_amdgcn_s_wait_asynccnt)
  __builtin_amdgcn_s_wait_asynccnt(0);
#else
  asm volatile("s_wait_asynccnt 0" ::: "memory");
#endif
#endif
}

// --- fragment loaders, wave32, contiguous k = 16*(l/16)+i in both A and B --------------

// A-matrix 16x32 f16 (MxK). lane l: M = l&15; lanes 0-15 hold K 0-15,
// lanes 16-31 hold K 16-31. Two contiguous 16B chunks -> ds_load_b128 x2.
__device__ __forceinline__ v16h load_a_frag(const _Float16* lds, int row0,
                                            int k0, int stride) {
  int lane = threadIdx.x & 31;
  int hi = lane >> 4, r = lane & 15;
  const _Float16* p = lds + (size_t)(row0 + r) * stride + k0;
  v16h a;
#pragma unroll
  for (int i = 0; i < 8; ++i) a[i] = p[hi * 16 + i];
#pragma unroll
  for (int i = 0; i < 8; ++i) a[8 + i] = p[hi * 16 + 8 + i];
  return a;
}

// B-matrix 32x16 f16 (KxN) where the logical B is the transpose of a
// row-major source S[n][k] kept in natural layout in LDS:
// lane l: n = l&15, koff = (l>>4)*16; b[i] = B[koff+i][n] = S[row0+n][k0+koff+i]
// -> 16 CONTIGUOUS halves per lane -> ds_load_b128 x2.
__device__ __forceinline__ v16h load_b_frag_rm(const _Float16* lds, int row0,
                                               int k0, int stride) {
  int lane = threadIdx.x & 31;
  int n = lane & 15, koff = (lane >> 4) * 16;
  const _Float16* p = lds + (size_t)(row0 + n) * stride + k0 + koff;
  v16h b;
#pragma unroll
  for (int i = 0; i < 16; ++i) b[i] = p[i];
  return b;
}

__device__ __forceinline__ v8f wmma_f16(v16h a, v16h b, v8f c) {
  return __builtin_amdgcn_wmma_f32_16x16x32_f16(false, a, false, b,
                                                (short)0, c, false, false);
}

// ---------------------------------------------------------------------------
// Kernel 1: projection GEMM. Block = 256 thr (8 waves), tile 64(M) x 128(F),
// K-step 32. Wave (wm,wf) owns a 32x32 output patch = 2x2 WMMA tiles.
// ---------------------------------------------------------------------------
__global__ __launch_bounds__(256)
void proj_qk_kernel(const float* __restrict__ x, const float* __restrict__ W,
                    const float* __restrict__ bias,
                    _Float16* __restrict__ qbuf, _Float16* __restrict__ kbuf) {
  __shared__ _Float16 Alds[64][40];    // x tile, natural [m][k], padded
  __shared__ _Float16 Wlds[128][40];   // W tile, natural [f][k], padded

  const int M0 = blockIdx.x * 64;
  const int F0 = blockIdx.y * 128;
  const int tid = threadIdx.x;
  const int wave = tid >> 5;
  const int wm = wave & 1;
  const int wf = wave >> 1;
  const int lane = tid & 31;
  const int colN = lane & 15, rgrp = lane >> 4;

  v8f acc[2][2];
#pragma unroll
  for (int i = 0; i < 2; ++i)
#pragma unroll
    for (int j = 0; j < 2; ++j) acc[i][j] = (v8f){};

  for (int k0 = 0; k0 < DIMC; k0 += 32) {
    __syncthreads();
#pragma unroll
    for (int t = 0; t < 2; ++t) {      // 64 rows x 8 float4
      int idx = tid + t * 256;
      int q4 = idx & 7, m = idx >> 3;
      float4 v = *(const float4*)&x[(size_t)(M0 + m) * DIMC + k0 + q4 * 4];
      *(v4h*)&Alds[m][q4 * 4] =
          (v4h){(_Float16)v.x, (_Float16)v.y, (_Float16)v.z, (_Float16)v.w};
    }
#pragma unroll
    for (int t = 0; t < 4; ++t) {      // 128 rows x 8 float4
      int idx = tid + t * 256;
      int q4 = idx & 7, f = idx >> 3;
      float4 v = *(const float4*)&W[(size_t)(F0 + f) * DIMC + k0 + q4 * 4];
      *(v4h*)&Wlds[f][q4 * 4] =
          (v4h){(_Float16)v.x, (_Float16)v.y, (_Float16)v.z, (_Float16)v.w};
    }
    __syncthreads();

#pragma unroll
    for (int mt = 0; mt < 2; ++mt) {
      v16h a = load_a_frag(&Alds[0][0], 32 * wm + 16 * mt, 0, 40);
#pragma unroll
      for (int ft = 0; ft < 2; ++ft) {
        v16h b = load_b_frag_rm(&Wlds[0][0], 32 * wf + 16 * ft, 0, 40);
        acc[mt][ft] = wmma_f16(a, b, acc[mt][ft]);
      }
    }
  }

  // epilogue: bias add, route to Q or K in [b,h,n,d] f16 layout
#pragma unroll
  for (int mt = 0; mt < 2; ++mt)
#pragma unroll
    for (int ft = 0; ft < 2; ++ft)
#pragma unroll
      for (int j = 0; j < 8; ++j) {
        int m = M0 + 32 * wm + 16 * mt + rgrp * 8 + j;
        int f = F0 + 32 * wf + 16 * ft + colN;
        float v = acc[mt][ft][j] + bias[f];
        int b = m >> 11;
        int n = m & (SEQN - 1);
        int fr = (f < DIMC) ? f : (f - DIMC);
        int h = fr >> 6, d = fr & 63;
        _Float16* dst = (f < DIMC) ? qbuf : kbuf;
        dst[(((size_t)(b * NHEAD + h)) * SEQN + n) * HDIM + d] = (_Float16)v;
      }
}

// ---------------------------------------------------------------------------
// Kernel 2: per-(b,h) streaming diag softmax. Block = 256 thr (8 waves),
// 64 queries per block; K streamed in 128-key chunks, async-to-LDS,
// double-buffered + software-pipelined. Wave (wq,wk): 32q x 32k.
// ---------------------------------------------------------------------------
__global__ __launch_bounds__(256)
void attn_diag_kernel(const _Float16* __restrict__ qbuf,
                      const _Float16* __restrict__ kbuf,
                      float* __restrict__ diag) {
  __shared__ _Float16 Qlds[64][72];        // 9.0 KB, natural [n][d]
  __shared__ _Float16 Klds[2][128][72];    // 36 KB, natural [key][d], 2 bufs
  __shared__ float Pm[64][4], Ps[64][4], Pd[64][4];

  const int bh = blockIdx.y;
  const int Q0 = blockIdx.x * 64;
  const size_t base = (size_t)bh * SEQN * HDIM;
  const int tid = threadIdx.x;
  const int wave = tid >> 5;
  const int wq = wave & 1;
  const int wk = wave >> 1;
  const int lane = tid & 31;
  const int colN = lane & 15, rgrp = lane >> 4;

  // --- chunk copy: contiguous 128x64 halves (16 KB) global -> LDS ----------
  auto stage_k = [&](int chunk, int buf) {
    const _Float16* src = kbuf + base + (size_t)chunk * 128 * HDIM;
#pragma unroll
    for (int t = 0; t < 4; ++t) {          // 1024 x 16B chunks / 256 thr
      int c = tid + t * 256;
      int key = c >> 3, d8 = c & 7;
#if ASYNC_LDS
      __builtin_amdgcn_global_load_async_to_lds_b128(
          (gv4i*)&src[(size_t)key * HDIM + d8 * 8],
          (lv4i*)&Klds[buf][key][d8 * 8], 0, 0);
#else
      *(v8h*)&Klds[buf][key][d8 * 8] =
          *(const v8h*)&src[(size_t)key * HDIM + d8 * 8];
#endif
    }
  };

  // prologue: stage chunk 0 + Q tile
  stage_k(0, 0);
#pragma unroll
  for (int t = 0; t < 4; ++t) {            // 64x64 Q tile, 16B chunks
    int c = tid + t * 256;
    int n = c >> 3, d8 = c & 7;
    *(v8h*)&Qlds[n][d8 * 8] = *(const v8h*)&qbuf[base + (size_t)(Q0 + n) * HDIM + d8 * 8];
  }
  wait_async0();
  __syncthreads();

  // Q fragments are chunk-invariant: hoist out of the key loop
  v16h aq[2][2];
#pragma unroll
  for (int mt = 0; mt < 2; ++mt) {
    aq[mt][0] = load_a_frag(&Qlds[0][0], 32 * wq + 16 * mt, 0, 72);
    aq[mt][1] = load_a_frag(&Qlds[0][0], 32 * wq + 16 * mt, 32, 72);
  }

  float rm[2][8], rs[2][8], rd[2][8];
#pragma unroll
  for (int mt = 0; mt < 2; ++mt)
#pragma unroll
    for (int j = 0; j < 8; ++j) { rm[mt][j] = -1e30f; rs[mt][j] = 0.f; rd[mt][j] = -1e30f; }

  for (int c = 0; c < SEQN / 128; ++c) {
    const int buf = c & 1;
    if (c + 1 < SEQN / 128) stage_k(c + 1, buf ^ 1);   // overlap with compute

    const _Float16* Kb = &Klds[buf][0][0];
#pragma unroll
    for (int mt = 0; mt < 2; ++mt) {
#pragma unroll
      for (int kt = 0; kt < 2; ++kt) {
        v16h b0 = load_b_frag_rm(Kb, 32 * wk + 16 * kt, 0, 72);
        v16h b1 = load_b_frag_rm(Kb, 32 * wk + 16 * kt, 32, 72);
        v8f sc = (v8f){};
        sc = wmma_f16(aq[mt][0], b0, sc);
        sc = wmma_f16(aq[mt][1], b1, sc);
        int colg = c * 128 + 32 * wk + 16 * kt + colN;
#pragma unroll
        for (int j = 0; j < 8; ++j) {
          float s = sc[j] * 0.125f;        // HEAD_DIM^-0.5
          int rowg = Q0 + 32 * wq + 16 * mt + rgrp * 8 + j;
          if (colg == rowg) rd[mt][j] = s;
          float mn = fmaxf(rm[mt][j], s);
          rs[mt][j] = rs[mt][j] * __expf(rm[mt][j] - mn) + __expf(s - mn);
          rm[mt][j] = mn;
        }
      }
    }
    wait_async0();
    __syncthreads();
  }

  // reduce 16 column-lanes per row group (width-16 xor stays in-group)
#pragma unroll
  for (int mt = 0; mt < 2; ++mt)
#pragma unroll
    for (int j = 0; j < 8; ++j) {
      float m = rm[mt][j], s = rs[mt][j], dg = rd[mt][j];
#pragma unroll
      for (int off = 1; off < 16; off <<= 1) {
        float m2 = __shfl_xor(m, off, 16);
        float s2 = __shfl_xor(s, off, 16);
        float d2 = __shfl_xor(dg, off, 16);
        float mn = fmaxf(m, m2);
        s = s * __expf(m - mn) + s2 * __expf(m2 - mn);
        m = mn;
        dg = fmaxf(dg, d2);
      }
      if (colN == 0) {
        int row = 32 * wq + 16 * mt + rgrp * 8 + j;
        Pm[row][wk] = m; Ps[row][wk] = s; Pd[row][wk] = dg;
      }
    }
  __syncthreads();

  if (tid < 64) {                          // combine the 4 key-wave partials
    float m = Pm[tid][0], s = Ps[tid][0], dg = Pd[tid][0];
#pragma unroll
    for (int t = 1; t < 4; ++t) {
      float m2 = Pm[tid][t], s2 = Ps[tid][t];
      float mn = fmaxf(m, m2);
      s = s * __expf(m - mn) + s2 * __expf(m2 - mn);
      m = mn;
      dg = fmaxf(dg, Pd[tid][t]);
    }
    diag[(size_t)bh * SEQN + Q0 + tid] = __expf(dg - m) / s;
  }
}

// ---------------------------------------------------------------------------
// Kernel 3: mean over heads -> out[B, N]
// ---------------------------------------------------------------------------
__global__ __launch_bounds__(256)
void head_mean_kernel(const float* __restrict__ diag, float* __restrict__ out) {
  int g = blockIdx.x * 256 + threadIdx.x;
  int b = g >> 11, n = g & (SEQN - 1);
  float s = 0.f;
#pragma unroll
  for (int h = 0; h < NHEAD; ++h)
    s += diag[((size_t)(b * NHEAD + h)) * SEQN + n];
  out[g] = s * (1.0f / NHEAD);
}

// ---------------------------------------------------------------------------
extern "C" void kernel_launch(void* const* d_in, const int* in_sizes, int n_in,
                              void* d_out, int out_size, void* d_ws, size_t ws_size,
                              hipStream_t stream) {
  const float* x    = (const float*)d_in[0];   // [4,2048,768]
  const float* W    = (const float*)d_in[1];   // [1536,768]
  const float* bias = (const float*)d_in[2];   // [1536]
  float* out = (float*)d_out;                  // [4,2048]

  char* ws = (char*)d_ws;
  const size_t qk_bytes = QK_HALFS * sizeof(_Float16);
  _Float16* qbuf = (_Float16*)ws;
  _Float16* kbuf = (_Float16*)(ws + qk_bytes);
  float* diag    = (float*)(ws + 2 * qk_bytes);

  proj_qk_kernel<<<dim3(MROWS / 64, FEATS / 128), 256, 0, stream>>>(
      x, W, bias, qbuf, kbuf);
  attn_diag_kernel<<<dim3(SEQN / 64, BATCH * NHEAD), 256, 0, stream>>>(
      qbuf, kbuf, diag);
  head_mean_kernel<<<dim3(BATCH * SEQN / 256), 256, 0, stream>>>(diag, out);
}
